// GraphModuleEmbedding_49667001811674
// MI455X (gfx1250) — matrix operand order repaired and checked by reading, not compiled, not run on hardware
//
#include <hip/hip_runtime.h>
#include <hip/hip_bf16.h>
#include <stdint.h>

// ---------------------------------------------------------------------------
// TGN-style graph embedding, MI455X (gfx1250, wave32, WMMA bf16 path)
//   GEMM1: [B*K, 384] x [384,128]  via v_wmma_f32_16x16x32_bf16
//   GEMM2: [B, 256]   x [256,128]  via v_wmma_f32_16x16x32_bf16
// ---------------------------------------------------------------------------

typedef __attribute__((ext_vector_type(16))) __bf16 v16bf;
typedef __attribute__((ext_vector_type(8)))  float  v8f;

#define B_   16384
#define K_   32
#define DN_  128
#define MSGP 392   // padded LDS row stride (bf16 elems): 784B = 196 dwords, conflict-free

__device__ __forceinline__ unsigned short f2bf(float f) {
  unsigned int u = __float_as_uint(f);
  u += 0x7FFFu + ((u >> 16) & 1u);          // round-to-nearest-even
  return (unsigned short)(u >> 16);
}
__device__ __forceinline__ unsigned int pack2bf(float a, float b) {
  return (unsigned int)f2bf(a) | ((unsigned int)f2bf(b) << 16);
}

union FragAB { uint4 u[2]; v16bf v; };

// ---------------------------------------------------------------------------
// Kernel 0: pack W1 (384x128 f32) and W2 (256x128 f32) into bf16 B-fragment
// order: per (ktile, ntile) a 32-lane x 8-dword block; lane l<16 -> col
// n=nt*16+l, rows k=kt*32+2j+(j>=4?8:0)..; lane>=16 -> same rows +8 (ISA 7.12.2).
// ---------------------------------------------------------------------------
__global__ void __launch_bounds__(256)
pack_weights(const float* __restrict__ W1, const float* __restrict__ W2,
             unsigned int* __restrict__ wp) {
  int t = blockIdx.x * 256 + threadIdx.x;
  const float* W;
  int rem;
  if (t < 24576) {            // W1: 12 ktiles * 8 ntiles * 256 dwords
    W = W1; rem = t;
  } else if (t < 40960) {     // W2: 8 ktiles * 8 ntiles * 256 dwords
    W = W2; rem = t - 24576;
  } else {
    return;
  }
  int kt   = rem / 2048;
  int nt   = (rem / 256) & 7;
  int lane = (rem / 8) & 31;
  int j    = rem & 7;
  int n  = nt * 16 + (lane & 15);
  int k0 = kt * 32 + ((lane & 16) ? 8 : 0) + 2 * j + ((j & 4) ? 8 : 0);
  wp[t] = pack2bf(W[(size_t)k0 * 128 + n], W[(size_t)(k0 + 1) * 128 + n]);
}

// ---------------------------------------------------------------------------
// Kernel 1: per batch b — gather nbr/edge rows, time encoding, stage 32x384
// bf16 message in LDS, GEMM1 (8 waves x [2 Mtiles x 12 ksteps] WMMAs),
// bias+ReLU+weighted K-reduction -> agg; emit x2[b] = bf16 concat(src, agg).
// ---------------------------------------------------------------------------
__global__ void __launch_bounds__(256)
tgn_gather_gemm1(const float* __restrict__ nodef, const float* __restrict__ edgef,
                 const int* __restrict__ seln, const int* __restrict__ sele,
                 const float* __restrict__ sdt, const float* __restrict__ sw,
                 const int* __restrict__ n0,
                 const float* __restrict__ tw, const float* __restrict__ tb,
                 const float* __restrict__ b1,
                 const uint4* __restrict__ w1pack,
                 unsigned short* __restrict__ x2) {
  __shared__ unsigned short msg[K_][MSGP];  // 25088 B
  __shared__ float warr[K_];
  __shared__ float wnorm[K_];

  const int tid = threadIdx.x;
  const int b   = blockIdx.x;

  // src row -> x2[b, 0:128] (bf16)
  if (tid < 128) {
    x2[(size_t)b * 256 + tid] = f2bf(nodef[(size_t)n0[b] * DN_ + tid]);
  }

  // ---- gather + bf16 convert into LDS: 8 threads per neighbor row ----
  const int k  = tid >> 3;         // 0..31
  const int cs = (tid & 7) * 16;   // 16-col segment
  const float* nrow = nodef + (size_t)seln[b * K_ + k] * DN_;
  const float* erow = edgef + (size_t)sele[b * K_ + k] * DN_;
  const float  dt   = sdt[b * K_ + k];
#pragma unroll
  for (int i = 0; i < 4; ++i) {
    float4 f = *(const float4*)(nrow + cs + i * 4);
    *(uint2*)&msg[k][cs + i * 4] = make_uint2(pack2bf(f.x, f.y), pack2bf(f.z, f.w));
    float4 g = *(const float4*)(erow + cs + i * 4);
    *(uint2*)&msg[k][128 + cs + i * 4] = make_uint2(pack2bf(g.x, g.y), pack2bf(g.z, g.w));
  }
#pragma unroll
  for (int c = 0; c < 16; c += 2) {
    float t0 = cosf(dt * tw[cs + c]     + tb[cs + c]);
    float t1 = cosf(dt * tw[cs + c + 1] + tb[cs + c + 1]);
    *(unsigned int*)&msg[k][256 + cs + c] = pack2bf(t0, t1);
  }
  if (tid < K_) warr[tid] = sw[b * K_ + tid];
  __syncthreads();

  // weight normalization (ws==0 -> all-zero weights, matches reference mask)
  if (tid == 0) {
    float s = 0.f;
    for (int i = 0; i < K_; ++i) s += warr[i];
    float inv = (s == 0.f) ? 0.f : 1.f / s;
    for (int i = 0; i < K_; ++i) wnorm[i] = warr[i] * inv;
  }

  // ---- GEMM1: wave = one 16-col N-tile; two M-tiles (K rows 0-15 / 16-31) ----
  const int wave  = tid >> 5;
  const int lane  = tid & 31;
  const int ntile = wave;          // 0..7
  const int half  = lane >> 4;     // A-frag half select (K 0-7 vs 8-15 etc.)
  const int mrow  = lane & 15;

  v8f c0 = {};  // rows 0..15  of H tile
  v8f c1 = {};  // rows 16..31 of H tile
  for (int kt = 0; kt < 12; ++kt) {
    const int co = kt * 32 + half * 8;
    FragAB a0, a1, bb;
    a0.u[0] = *(const uint4*)&msg[mrow][co];
    a0.u[1] = *(const uint4*)&msg[mrow][co + 16];
    a1.u[0] = *(const uint4*)&msg[16 + mrow][co];
    a1.u[1] = *(const uint4*)&msg[16 + mrow][co + 16];
    const uint4* bp = w1pack + ((size_t)(kt * 8 + ntile) * 32 + lane) * 2;
    bb.u[0] = bp[0];
    bb.u[1] = bp[1];
    c0 = __builtin_amdgcn_wmma_f32_16x16x32_bf16(false, a0.v, false, bb.v,
                                                 (short)0, c0, false, false);
    c1 = __builtin_amdgcn_wmma_f32_16x16x32_bf16(false, a1.v, false, bb.v,
                                                 (short)0, c1, false, false);
  }
  __syncthreads();   // wnorm visible to all epilogue readers

  // ---- epilogue: bias + ReLU + weighted reduce over K=32 neighbors ----
  const int n    = ntile * 16 + mrow;
  const float bn = b1[n];
  float p = 0.f;
#pragma unroll
  for (int r = 0; r < 8; ++r) {
    const int m0 = r + half * 8;       // C layout: lanes>=16 hold rows +8
    float h0 = c0[r] + bn; h0 = h0 > 0.f ? h0 : 0.f;
    float h1 = c1[r] + bn; h1 = h1 > 0.f ? h1 : 0.f;
    p += h0 * wnorm[m0] + h1 * wnorm[16 + m0];
  }
  p += __shfl_xor(p, 16, 32);          // combine the two lane-halves
  if (lane < 16) x2[(size_t)b * 256 + 128 + n] = f2bf(p);
}

// ---------------------------------------------------------------------------
// Kernel 2: out = relu(x2[B,256] @ W2 + b2), f32 output [B,128]
// One wave per (mtile, ntile); 8 k-steps of WMMA.
// ---------------------------------------------------------------------------
__global__ void __launch_bounds__(256)
tgn_gemm2(const unsigned short* __restrict__ x2,
          const uint4* __restrict__ w2pack,
          const float* __restrict__ b2,
          float* __restrict__ out) {
  const int tid   = threadIdx.x;
  const int wave  = tid >> 5;
  const int lane  = tid & 31;
  const int gid   = blockIdx.x * 8 + wave;
  const int mtile = gid >> 3;     // 0..1023
  const int ntile = gid & 7;      // 0..7
  const int half  = lane >> 4;
  const int mrow  = lane & 15;

  const unsigned short* arow = x2 + (size_t)(mtile * 16 + mrow) * 256;
  v8f c = {};
  for (int kt = 0; kt < 8; ++kt) {
    const int co = kt * 32 + half * 8;
    FragAB a, bb;
    a.u[0] = *(const uint4*)(arow + co);
    a.u[1] = *(const uint4*)(arow + co + 16);
    const uint4* bp = w2pack + ((size_t)(kt * 8 + ntile) * 32 + lane) * 2;
    bb.u[0] = bp[0];
    bb.u[1] = bp[1];
    c = __builtin_amdgcn_wmma_f32_16x16x32_bf16(false, a.v, false, bb.v,
                                                (short)0, c, false, false);
  }
  const int n    = ntile * 16 + mrow;
  const float bn = b2[n];
#pragma unroll
  for (int r = 0; r < 8; ++r) {
    float v = c[r] + bn; v = v > 0.f ? v : 0.f;
    out[(size_t)(mtile * 16 + r + half * 8) * 128 + n] = v;
  }
}

// ---------------------------------------------------------------------------
extern "C" void kernel_launch(void* const* d_in, const int* in_sizes, int n_in,
                              void* d_out, int out_size, void* d_ws, size_t ws_size,
                              hipStream_t stream) {
  const float* nodef = (const float*)d_in[0];   // [500000,128]
  const float* edgef = (const float*)d_in[1];   // [2000000,128]
  const int*   seln  = (const int*)d_in[2];     // [B,K]
  const int*   sele  = (const int*)d_in[3];     // [B,K]
  const float* sdt   = (const float*)d_in[4];   // [B,K]
  const float* sw    = (const float*)d_in[5];   // [B,K]
  const int*   n0    = (const int*)d_in[6];     // [B]
  const float* tw    = (const float*)d_in[7];   // [128]
  const float* tb    = (const float*)d_in[8];   // [128]
  const float* W1    = (const float*)d_in[9];   // [384,128]
  const float* b1    = (const float*)d_in[10];  // [128]
  const float* W2    = (const float*)d_in[11];  // [256,128]
  const float* b2    = (const float*)d_in[12];  // [128]

  // workspace layout: W1 pack (98304 B) | W2 pack (65536 B) | x2 bf16 (8 MiB)
  unsigned int*   wp     = (unsigned int*)d_ws;
  const uint4*    w1pack = (const uint4*)d_ws;
  const uint4*    w2pack = (const uint4*)((char*)d_ws + 98304);
  unsigned short* x2     = (unsigned short*)((char*)d_ws + 163840);

  pack_weights<<<160, 256, 0, stream>>>(W1, W2, wp);
  tgn_gather_gemm1<<<B_, 256, 0, stream>>>(nodef, edgef, seln, sele, sdt, sw,
                                           n0, tw, tb, b1, w1pack, x2);
  tgn_gemm2<<<B_ / 16, 256, 0, stream>>>(x2, w2pack, b2, (float*)d_out);
}
